// C_VAE_NET_56590489092414
// MI455X (gfx1250) — matrix operand order repaired and checked by reading, not compiled, not run on hardware
//
#include <hip/hip_runtime.h>

// ---------------------------------------------------------------------------
// Types for CDNA5 WMMA (wave32): v_wmma_f32_16x16x32_bf16
// ---------------------------------------------------------------------------
typedef unsigned short ushort_t;
typedef __attribute__((ext_vector_type(16))) __bf16 v16bf;
typedef __attribute__((ext_vector_type(8)))  float  v8f;

union FragBF {
    unsigned int u[8];
    uint4        q[2];
    v16bf        v;
};

__device__ __forceinline__ ushort_t f2bf(float f) {
    union { float f; unsigned int u; } cv; cv.f = f;
    unsigned int u = cv.u + 0x7FFFu + ((cv.u >> 16) & 1u);  // RNE
    return (ushort_t)(u >> 16);
}

// A fragment: 16x32 bf16, row-major source, row stride `ld` elements.
// ISA layout: lane l -> M = l%16, h = l/16; VGPR0-3 = K[kb+8h .. kb+8h+7],
// VGPR4-7 = K[kb+16+8h .. kb+16+8h+7].
__device__ __forceinline__ v16bf load_a_frag(const ushort_t* base, int ld,
                                             int kbase, int lane) {
    int m = lane & 15;
    int h = (lane >> 4) & 1;
    const ushort_t* p = base + (size_t)m * ld + kbase + h * 8;
    FragBF f;
    f.q[0] = *(const uint4*)p;
    f.q[1] = *(const uint4*)(p + 16);
    return f.v;
}

// B fragment: 32x16 bf16 (KxN) sourced from pre-transposed W^T [N][K].
// ISA layout: lane l -> N = l%16, h = l/16; VGPR0-3 = K[kb+8h .. kb+8h+7],
// VGPR4-7 = K[kb+16+8h .. kb+16+8h+7] (same K order as the A fragment).
__device__ __forceinline__ v16bf load_b_frag(const ushort_t* wt, int ldk,
                                             int nbase, int kbase, int lane) {
    int n = lane & 15;
    int h = (lane >> 4) & 1;
    const ushort_t* p = wt + (size_t)(nbase + n) * ldk + kbase + h * 8;
    FragBF f;
    f.q[0] = *(const uint4*)p;
    f.q[1] = *(const uint4*)(p + 16);
    return f.v;
}

#define WMMA_BF16(a, b, c) \
    __builtin_amdgcn_wmma_f32_16x16x32_bf16(false, (a), false, (b), (short)0, (c), false, false)

// ---------------------------------------------------------------------------
// fp32 -> bf16 conversion
// ---------------------------------------------------------------------------
__global__ __launch_bounds__(256) void vae_cvt_bf16(const float* __restrict__ in,
                                                    ushort_t* __restrict__ out,
                                                    int n) {
    int i = blockIdx.x * 256 + threadIdx.x;
    if (i < n) out[i] = f2bf(in[i]);
}

// Pack [Wx;Wh] (each stored [K][N] row-major fp32) into bf16 W^T [N][Kcat].
// For plain dense weights pass Wh=nullptr and Din=Kcat.
__global__ __launch_bounds__(256) void vae_pack_wt(const float* __restrict__ Wx,
                                                   const float* __restrict__ Wh,
                                                   ushort_t* __restrict__ out,
                                                   int N, int Din, int Kcat) {
    int idx = blockIdx.x * 256 + threadIdx.x;
    if (idx >= N * Kcat) return;
    int n = idx / Kcat;
    int k = idx - n * Kcat;
    float v = (k < Din) ? Wx[(size_t)k * N + n] : Wh[(size_t)(k - Din) * N + n];
    out[idx] = f2bf(v);
}

// ---------------------------------------------------------------------------
// LSTM scan: one workgroup = 16 batch rows, full T-step recurrence.
// Wave w owns gate columns u in [w*64, w*64+64): for each 16-col u-tile it
// accumulates the i/f/g/o tiles (u, u+U, u+2U, u+3U) with 4 WMMAs per shared
// A fragment, then does the gate nonlinearities fully in registers.
// LDS (double buffered, one barrier/step):
//   X0[16][Din] | X1[16][Din] | H0[16][U] | H1[16][U]   (<= 96 KB)
// Cell state c lives in 32 VGPRs per lane.
// ---------------------------------------------------------------------------
__global__ __launch_bounds__(256) void vae_lstm_scan(
    const ushort_t* __restrict__ xin,   // [B][T][Din] bf16
    const ushort_t* __restrict__ WT,    // [4U][Din+U] bf16
    const float*    __restrict__ bias,  // [4U]
    ushort_t*       __restrict__ hout,  // [B][T][U] bf16
    int T, int Din) {
    const int U = 512;
    extern __shared__ char smem[];
    const int Kcat = Din + U;
    ushort_t* Xsh0 = (ushort_t*)smem;
    ushort_t* Xsh1 = Xsh0 + 16 * Din;
    ushort_t* Hsh0 = Xsh1 + 16 * Din;
    ushort_t* Hsh1 = Hsh0 + 16 * U;

    const int tid   = threadIdx.x;
    const int lane  = tid & 31;
    const int wave  = tid >> 5;          // 0..7
    const int bbase = blockIdx.x * 16;
    const int n     = lane & 15;
    const int hh    = (lane >> 4) & 1;

    // zero initial hidden buffer 0 (read at t=0)
    for (int i = tid; i < 16 * U; i += 256) Hsh0[i] = (ushort_t)0;

    // hoist per-column gate biases out of the time loop
    float bg[4][4];
#pragma unroll
    for (int ut = 0; ut < 4; ++ut) {
        int u0 = wave * 64 + ut * 16;
#pragma unroll
        for (int g = 0; g < 4; ++g) bg[ut][g] = bias[u0 + g * U + n];
    }

    // cell state in registers: 4 u-tiles x 8 rows per lane
    float creg[4][8];
#pragma unroll
    for (int ut = 0; ut < 4; ++ut)
#pragma unroll
        for (int r = 0; r < 8; ++r) creg[ut][r] = 0.0f;

    const v8f vzero = {0.f, 0.f, 0.f, 0.f, 0.f, 0.f, 0.f, 0.f};

    for (int t = 0; t < T; ++t) {
        ushort_t*       Xcur = (t & 1) ? Xsh1 : Xsh0;
        const ushort_t* Hcur = (t & 1) ? Hsh1 : Hsh0;
        ushort_t*       Hnxt = (t & 1) ? Hsh0 : Hsh1;

        // stage x_t tile (double buffered -> no trailing barrier needed)
        for (int i = tid; i < (16 * Din) / 4; i += 256) {
            int e = i * 4;
            int m = e / Din, col = e - m * Din;
            *(uint2*)(Xcur + m * Din + col) =
                *(const uint2*)(xin + ((size_t)(bbase + m) * T + t) * Din + col);
        }
        __syncthreads();

#pragma unroll
        for (int ut = 0; ut < 4; ++ut) {
            int u0 = wave * 64 + ut * 16;
            v8f ai = vzero, af = vzero, ag = vzero, ao = vzero;

            // K over x-part (Din is a multiple of 32)
            for (int k = 0; k < Din; k += 32) {
                v16bf a  = load_a_frag(Xcur, Din, k, lane);
                v16bf b0 = load_b_frag(WT, Kcat, u0,         k, lane);
                v16bf b1 = load_b_frag(WT, Kcat, u0 + U,     k, lane);
                v16bf b2 = load_b_frag(WT, Kcat, u0 + 2 * U, k, lane);
                v16bf b3 = load_b_frag(WT, Kcat, u0 + 3 * U, k, lane);
                ai = WMMA_BF16(a, b0, ai);
                af = WMMA_BF16(a, b1, af);
                ag = WMMA_BF16(a, b2, ag);
                ao = WMMA_BF16(a, b3, ao);
            }
            // K over h-part
            for (int k = 0; k < U; k += 32) {
                v16bf a  = load_a_frag(Hcur, U, k, lane);
                v16bf b0 = load_b_frag(WT, Kcat, u0,         Din + k, lane);
                v16bf b1 = load_b_frag(WT, Kcat, u0 + U,     Din + k, lane);
                v16bf b2 = load_b_frag(WT, Kcat, u0 + 2 * U, Din + k, lane);
                v16bf b3 = load_b_frag(WT, Kcat, u0 + 3 * U, Din + k, lane);
                ai = WMMA_BF16(a, b0, ai);
                af = WMMA_BF16(a, b1, af);
                ag = WMMA_BF16(a, b2, ag);
                ao = WMMA_BF16(a, b3, ao);
            }

            // gate nonlinearities + state update, all in registers
#pragma unroll
            for (int r = 0; r < 8; ++r) {
                float iv = 1.0f / (1.0f + __expf(-(ai[r] + bg[ut][0])));
                float fv = 1.0f / (1.0f + __expf(-(af[r] + bg[ut][1])));
                float gv = tanhf(ag[r] + bg[ut][2]);
                float ov = 1.0f / (1.0f + __expf(-(ao[r] + bg[ut][3])));
                float c  = fv * creg[ut][r] + iv * gv;
                creg[ut][r] = c;
                ushort_t hb = f2bf(ov * tanhf(c));
                int m = r + 8 * hh;
                Hnxt[m * U + u0 + n] = hb;                            // next-step A
                hout[((size_t)(bbase + m) * T + t) * U + u0 + n] = hb; // sequence out
            }
        }
        // no trailing barrier: X and H are double buffered; the loop-top
        // barrier at t+1 orders this step's Hnxt writes before their reads.
    }
}

// ---------------------------------------------------------------------------
// Dense: out[M][N] = A_bf16[M][K] @ W^T_bf16[N][K] + bias, fp32 out.
// One wave per 16x64 output block: 4 accumulators share each A fragment.
// ---------------------------------------------------------------------------
__global__ __launch_bounds__(256) void vae_dense_wmma(
    const ushort_t* __restrict__ A, const ushort_t* __restrict__ WT,
    const float* __restrict__ bias, float* __restrict__ out,
    int M, int K, int N) {
    int wid  = blockIdx.x * 8 + (threadIdx.x >> 5);
    int lane = threadIdx.x & 31;
    int nb4  = N >> 6;                      // number of 64-col blocks
    if (wid >= (M >> 4) * nb4) return;      // wave-uniform exit
    int mt = wid / nb4, nb = wid - mt * nb4;
    int mbase = mt << 4, nbase = nb << 6;

    const ushort_t* arow = A + (size_t)mbase * K;
    const v8f vzero = {0.f, 0.f, 0.f, 0.f, 0.f, 0.f, 0.f, 0.f};
    v8f acc[4];
#pragma unroll
    for (int j = 0; j < 4; ++j) acc[j] = vzero;

    for (int k = 0; k < K; k += 32) {
        v16bf a = load_a_frag(arow, K, k, lane);
#pragma unroll
        for (int j = 0; j < 4; ++j) {
            v16bf b = load_b_frag(WT, K, nbase + j * 16, k, lane);
            acc[j] = WMMA_BF16(a, b, acc[j]);
        }
    }
    int n = lane & 15, hh = (lane >> 4) & 1;
#pragma unroll
    for (int j = 0; j < 4; ++j) {
        float bv = bias[nbase + j * 16 + n];
#pragma unroll
        for (int r = 0; r < 8; ++r)
            out[(size_t)(mbase + r + 8 * hh) * N + nbase + j * 16 + n] = acc[j][r] + bv;
    }
}

// ---------------------------------------------------------------------------
// Reparameterize: z = mu + exp(0.5*logvar)*eps; also emits exp(logvar), z_bf16
// ---------------------------------------------------------------------------
__global__ __launch_bounds__(256) void vae_reparam(
    const float* __restrict__ zm, const float* __restrict__ zlv,
    const float* __restrict__ eps, float* __restrict__ z_out,
    float* __restrict__ exp_out, ushort_t* __restrict__ z_bf, int n) {
    int i = blockIdx.x * 256 + threadIdx.x;
    if (i < n) {
        float m = zm[i], lv = zlv[i];
        float zv = m + __expf(0.5f * lv) * eps[i];
        z_out[i]   = zv;
        exp_out[i] = __expf(lv);
        z_bf[i]    = f2bf(zv);
    }
}

// ---------------------------------------------------------------------------
extern "C" void kernel_launch(void* const* d_in, const int* in_sizes, int n_in,
                              void* d_out, int out_size, void* d_ws, size_t ws_size,
                              hipStream_t stream) {
    const int B = 512, T = 128, D = 128, Z = 64, U = 512;
    const int Gg = 4 * U;  // 2048 gate columns

    const float* x     = (const float*)d_in[0];
    const float* eps   = (const float*)d_in[1];
    const float* e0_Wx = (const float*)d_in[2];
    const float* e0_Wh = (const float*)d_in[3];
    const float* e0_b  = (const float*)d_in[4];
    const float* e1_Wx = (const float*)d_in[5];
    const float* e1_Wh = (const float*)d_in[6];
    const float* e1_b  = (const float*)d_in[7];
    const float* d0_Wx = (const float*)d_in[8];
    const float* d0_Wh = (const float*)d_in[9];
    const float* d0_b  = (const float*)d_in[10];
    const float* d1_Wx = (const float*)d_in[11];
    const float* d1_Wh = (const float*)d_in[12];
    const float* d1_b  = (const float*)d_in[13];
    const float* Wm    = (const float*)d_in[14];
    const float* bm    = (const float*)d_in[15];
    const float* Wv    = (const float*)d_in[16];
    const float* bv    = (const float*)d_in[17];
    const float* Wo    = (const float*)d_in[18];
    const float* bo    = (const float*)d_in[19];

    // workspace bump allocator (256B aligned)
    char*  ws  = (char*)d_ws;
    size_t off = 0;
    auto alloc = [&](size_t bytes) -> void* {
        void* p = ws + off;
        off += (bytes + 255) & ~(size_t)255;
        return p;
    };
    ushort_t* x_bf = (ushort_t*)alloc((size_t)B * T * D * 2);
    ushort_t* hA   = (ushort_t*)alloc((size_t)B * T * U * 2);  // h1 / g1
    ushort_t* hB   = (ushort_t*)alloc((size_t)B * T * U * 2);  // h2 / g2
    ushort_t* z_bf = (ushort_t*)alloc((size_t)B * T * Z * 2);
    ushort_t* We0  = (ushort_t*)alloc((size_t)Gg * (D + U) * 2);
    ushort_t* We1  = (ushort_t*)alloc((size_t)Gg * (U + U) * 2);
    ushort_t* Wd0  = (ushort_t*)alloc((size_t)Gg * (Z + U) * 2);
    ushort_t* Wd1  = (ushort_t*)alloc((size_t)Gg * (U + U) * 2);
    ushort_t* WmT  = (ushort_t*)alloc((size_t)Z * U * 2);
    ushort_t* WvT  = (ushort_t*)alloc((size_t)Z * U * 2);
    ushort_t* WoT  = (ushort_t*)alloc((size_t)D * U * 2);

    // output slices, concatenated in reference return order
    float* out    = (float*)d_out;
    float* o_rec  = out;
    float* o_exp  = o_rec + (size_t)B * T * D;
    float* o_mean = o_exp + (size_t)B * T * Z;
    float* o_lvar = o_mean + (size_t)B * T * Z;
    float* o_z    = o_lvar + (size_t)B * T * Z;

    // 1) convert x, pack all weights to bf16 W^T
    { int n = B * T * D;    vae_cvt_bf16<<<(n + 255) / 256, 256, 0, stream>>>(x, x_bf, n); }
    { int n = Gg * (D + U); vae_pack_wt<<<(n + 255) / 256, 256, 0, stream>>>(e0_Wx, e0_Wh, We0, Gg, D, D + U); }
    { int n = Gg * (U + U); vae_pack_wt<<<(n + 255) / 256, 256, 0, stream>>>(e1_Wx, e1_Wh, We1, Gg, U, U + U); }
    { int n = Gg * (Z + U); vae_pack_wt<<<(n + 255) / 256, 256, 0, stream>>>(d0_Wx, d0_Wh, Wd0, Gg, Z, Z + U); }
    { int n = Gg * (U + U); vae_pack_wt<<<(n + 255) / 256, 256, 0, stream>>>(d1_Wx, d1_Wh, Wd1, Gg, U, U + U); }
    { int n = Z * U;        vae_pack_wt<<<(n + 255) / 256, 256, 0, stream>>>(Wm, nullptr, WmT, Z, U, U); }
    { int n = Z * U;        vae_pack_wt<<<(n + 255) / 256, 256, 0, stream>>>(Wv, nullptr, WvT, Z, U, U); }
    { int n = D * U;        vae_pack_wt<<<(n + 255) / 256, 256, 0, stream>>>(Wo, nullptr, WoT, D, U, U); }

    // shared mem: 2*X + 2*H bf16 buffers
    auto scan_shmem = [&](int Din) -> size_t {
        return (size_t)(2 * 16 * Din + 2 * 16 * U) * 2;
    };

    // 2) encoder scans
    vae_lstm_scan<<<B / 16, 256, scan_shmem(D), stream>>>(x_bf, We0, e0_b, hA, T, D);
    vae_lstm_scan<<<B / 16, 256, scan_shmem(U), stream>>>(hA, We1, e1_b, hB, T, U);

    // 3) latent heads + reparameterize
    { int waves = (B * T / 16) * (Z / 64);
      vae_dense_wmma<<<(waves + 7) / 8, 256, 0, stream>>>(hB, WmT, bm, o_mean, B * T, U, Z); }
    { int waves = (B * T / 16) * (Z / 64);
      vae_dense_wmma<<<(waves + 7) / 8, 256, 0, stream>>>(hB, WvT, bv, o_lvar, B * T, U, Z); }
    { int n = B * T * Z;
      vae_reparam<<<(n + 255) / 256, 256, 0, stream>>>(o_mean, o_lvar, eps, o_z, o_exp, z_bf, n); }

    // 4) decoder scans + reconstruction head
    vae_lstm_scan<<<B / 16, 256, scan_shmem(Z), stream>>>(z_bf, Wd0, d0_b, hA, T, Z);
    vae_lstm_scan<<<B / 16, 256, scan_shmem(U), stream>>>(hA, Wd1, d1_b, hB, T, U);
    { int waves = (B * T / 16) * (D / 64);
      vae_dense_wmma<<<(waves + 7) / 8, 256, 0, stream>>>(hB, WoT, bo, o_rec, B * T, U, D); }
}
